// VolRender_2800318677677
// MI455X (gfx1250) — matrix operand order repaired and checked by reading, not compiled, not run on hardware
//
#include <hip/hip_runtime.h>
#include <hip/hip_bf16.h>

typedef _Float16 v16h __attribute__((ext_vector_type(16)));
typedef float    v8f  __attribute__((ext_vector_type(8)));

#define IMGSZ 128
#define HWH   64
#define NPTS  64
#define DV    64
#define NB    8
#define CF    16

__device__ inline v8f wmma16(v16h a, v16h b, v8f c) {
    // D = A(16x32 f16) x B(32x16 f16) + C(16x16 f32)
    return __builtin_amdgcn_wmma_f32_16x16x32_f16(
        /*neg_a=*/false, a, /*neg_b=*/false, b,
        /*c_mod=*/(short)0, c, /*reuse_a=*/false, /*reuse_b=*/false);
}

// ---------------------------------------------------------------------------
// Kernel 1: ray-march + trilinear sample + emission-absorption compositing.
// Output img is CHANNEL-LAST (b, y, x, 16) so conv B-fragments load 64B/lane.
// ---------------------------------------------------------------------------
__global__ void vr_render(const float* __restrict__ R, const float* __restrict__ T,
                          const float* __restrict__ Kc, const float* __restrict__ feat,
                          const float* __restrict__ dens, float* __restrict__ img,
                          float* __restrict__ sil)
{
    int tid = blockIdx.x * blockDim.x + threadIdx.x;
    if (tid >= NB * HWH * HWH) return;
    int b = tid >> 12;
    int rem = tid & 4095;
    int py = rem >> 6, px = rem & 63;

    const float* Rb = R + b * 9;
    const float* Tb = T + b * 3;
    const float* Kb = Kc + b * 9;
    float fxk = Kb[0] * 0.5f, fyk = Kb[4] * 0.5f;
    float cxk = Kb[2] * 0.5f, cyk = Kb[5] * 0.5f;
    float dcx = ((float)px + 0.5f - cxk) / fxk;
    float dcy = ((float)py + 0.5f - cyk) / fyk;
    float dwx = Rb[0]*dcx + Rb[3]*dcy + Rb[6];
    float dwy = Rb[1]*dcx + Rb[4]*dcy + Rb[7];
    float dwz = Rb[2]*dcx + Rb[5]*dcy + Rb[8];
    float orx = -(Rb[0]*Tb[0] + Rb[3]*Tb[1] + Rb[6]*Tb[2]);
    float ory = -(Rb[1]*Tb[0] + Rb[4]*Tb[1] + Rb[7]*Tb[2]);
    float orz = -(Rb[2]*Tb[0] + Rb[5]*Tb[1] + Rb[8]*Tb[2]);

    const float scale = 2.0f * (float)DV / (float)(DV - 1);
    const float* fb = feat + (size_t)b * CF * DV * DV * DV;
    const float* db = dens + (size_t)b * DV * DV * DV;

    float acc[CF];
    #pragma unroll
    for (int c = 0; c < CF; c++) acc[c] = 0.f;
    float absorb = 1.f, prod = 1.f;

    for (int p = 0; p < NPTS; p++) {
        float d = 1.2f + (1.6f / 63.f) * (float)p;
        float ixf = ((orx + dwx * d) * scale + 1.f) * 0.5f * (float)(DV - 1);
        float iyf = ((ory + dwy * d) * scale + 1.f) * 0.5f * (float)(DV - 1);
        float izf = ((orz + dwz * d) * scale + 1.f) * 0.5f * (float)(DV - 1);
        float x0f = floorf(ixf), y0f = floorf(iyf), z0f = floorf(izf);
        float fxr = ixf - x0f, fyr = iyf - y0f, fzr = izf - z0f;
        int x0 = (int)x0f, y0 = (int)y0f, z0 = (int)z0f;

        float cw[8]; int cidx[8];
        #pragma unroll
        for (int k = 0; k < 8; k++) {
            int dz = k >> 2, dy = (k >> 1) & 1, dx = k & 1;
            int xc = x0 + dx, yc = y0 + dy, zc = z0 + dz;
            float w = (dx ? fxr : 1.f - fxr) * (dy ? fyr : 1.f - fyr) * (dz ? fzr : 1.f - fzr);
            bool valid = (xc >= 0) && (xc < DV) && (yc >= 0) && (yc < DV) && (zc >= 0) && (zc < DV);
            int xcc = min(max(xc, 0), DV - 1);
            int ycc = min(max(yc, 0), DV - 1);
            int zcc = min(max(zc, 0), DV - 1);
            cw[k] = valid ? w : 0.f;
            cidx[k] = (zcc * DV + ycc) * DV + xcc;
        }
        float sigma = 0.f;
        #pragma unroll
        for (int k = 0; k < 8; k++) sigma += cw[k] * db[cidx[k]];
        float wgt = sigma * absorb;
        absorb *= (1.f + 1e-10f - sigma);
        prod   *= (1.f - sigma);
        for (int c = 0; c < CF; c++) {
            const float* fc = fb + (size_t)c * DV * DV * DV;
            float fv = 0.f;
            #pragma unroll
            for (int k = 0; k < 8; k++) fv += cw[k] * fc[cidx[k]];
            acc[c] += wgt * fv;
        }
    }
    float4* dst = (float4*)(img + (((size_t)b * HWH + py) * HWH + px) * CF);
    dst[0] = make_float4(acc[0],  acc[1],  acc[2],  acc[3]);
    dst[1] = make_float4(acc[4],  acc[5],  acc[6],  acc[7]);
    dst[2] = make_float4(acc[8],  acc[9],  acc[10], acc[11]);
    dst[3] = make_float4(acc[12], acc[13], acc[14], acc[15]);
    sil[((size_t)b * HWH + py) * HWH + px] = 1.f - prod;
}

// ---------------------------------------------------------------------------
// Weight pre-pack kernels: lay weights out in contiguous-k per-lane A-fragment
// order so conv waves fetch A with one 32B v16h load (2x b128).
// A layout: lane%16 = M(out ch); K(j) = j + 16*(lane>=16).
// ---------------------------------------------------------------------------
__global__ void vr_pack_wt(const float* __restrict__ wt, _Float16* __restrict__ aT)
{
    int tid = blockIdx.x * blockDim.x + threadIdx.x;   // par*160 + c*32 + lane
    if (tid >= 4 * 5 * 32) return;
    int lane = tid & 31;
    int c = (tid >> 5) % 5;
    int par = tid / 160;
    int py = par >> 1, px = par & 1;
    int hi = lane >> 4, o = lane & 15;
    #pragma unroll
    for (int j = 0; j < 16; j++) {
        int Kk = j + 16 * hi;
        int t = 2 * c + (Kk >> 4);
        int ich = Kk & 15;
        float wv = 0.f;
        if (t < 9) {
            int r = t / 3, s = t - 3 * r;
            int kh = 2 * r + 1 - py, kw = 2 * s + 1 - px;
            wv = wt[((ich * 16 + o) * 6 + (5 - kh)) * 6 + (5 - kw)];
        }
        aT[(size_t)tid * 16 + j] = (_Float16)wv;
    }
}

__global__ void vr_pack_w1(const float* __restrict__ w1, _Float16* __restrict__ aP)
{
    int tid = blockIdx.x * blockDim.x + threadIdx.x;   // c*32 + lane
    if (tid >= 13 * 32) return;
    int lane = tid & 31;
    int c = tid >> 5;
    int hi = lane >> 4, o = lane & 15;
    #pragma unroll
    for (int j = 0; j < 16; j++) {
        int Kk = j + 16 * hi;
        int t = 2 * c + (Kk >> 4);
        int ich = Kk & 15;
        float wv = ((o < 8) && (t < 25)) ? w1[(o * 16 + ich) * 25 + t] : 0.f;
        aP[(size_t)tid * 16 + j] = (_Float16)wv;
    }
}

__global__ void vr_pack_w2(const float* __restrict__ w2, _Float16* __restrict__ aP)
{
    int tid = blockIdx.x * blockDim.x + threadIdx.x;   // c*32 + lane
    if (tid >= 7 * 32) return;
    int lane = tid & 31;
    int c = tid >> 5;
    int hi = lane >> 4, o = lane & 15;
    #pragma unroll
    for (int j = 0; j < 16; j++) {
        int Kk = j + 16 * hi;
        int t = 4 * c + (Kk >> 3);
        int ich = Kk & 7;
        float wv = ((o < 3) && (t < 25)) ? w2[(o * 8 + ich) * 25 + t] : 0.f;
        aP[(size_t)tid * 16 + j] = (_Float16)wv;
    }
}

// ---------------------------------------------------------------------------
// Kernel 2: ConvTranspose 16->16 (6x6, stride 2) as 4 parity-class 3x3 convs.
// A = packed v16h load; B = 4x b128 + branch-free masking.
// ---------------------------------------------------------------------------
__global__ void vr_convt(const float* __restrict__ img, const _Float16* __restrict__ aT,
                         const float* __restrict__ bt, float* __restrict__ y1)
{
    int wave = (blockIdx.x * blockDim.x + threadIdx.x) >> 5;
    int lane = threadIdx.x & 31;
    int b = wave >> 10;
    int rem = wave & 1023;
    int py = (rem >> 9) & 1, px = (rem >> 8) & 1;
    int rem2 = rem & 255;
    int my = rem2 >> 2, tx = rem2 & 3;
    int hi = lane >> 4;
    int n  = lane & 15;
    int mx = tx * 16 + n;
    int par = py * 2 + px;

    const v16h* aFrag = (const v16h*)(aT + (size_t)(par * 5) * 32 * 16) + lane;

    v8f acc = {0.f,0.f,0.f,0.f,0.f,0.f,0.f,0.f};
    #pragma unroll
    for (int c = 0; c < 5; c++) {
        v16h a = aFrag[c * 32];
        // B fragment: lo lanes tap 2c, hi lanes tap 2c+1; clamped + masked loads
        int t = 2 * c + hi;
        int tc_ = (t < 9) ? t : 8;
        int r = tc_ / 3, s = tc_ - 3 * r;
        int iy = my + r - 1, ix = mx + s - 1;
        float msk = ((t < 9) && (iy >= 0) && (iy < HWH) && (ix >= 0) && (ix < HWH)) ? 1.f : 0.f;
        int iyc = min(max(iy, 0), HWH - 1);
        int ixc = min(max(ix, 0), HWH - 1);
        const float4* p = (const float4*)(img + (((size_t)b * HWH + iyc) * HWH + ixc) * CF);
        float4 q0 = p[0], q1 = p[1], q2 = p[2], q3 = p[3];
        float tmp[16] = {q0.x,q0.y,q0.z,q0.w, q1.x,q1.y,q1.z,q1.w,
                         q2.x,q2.y,q2.z,q2.w, q3.x,q3.y,q3.z,q3.w};
        v16h bb;
        #pragma unroll
        for (int j = 0; j < 16; j++) bb[j] = (_Float16)(tmp[j] * msk);
        acc = wmma16(a, bb, acc);
    }
    int oy = 2 * my + py;
    int ox = 2 * mx + px;
    float st[8];
    #pragma unroll
    for (int r8 = 0; r8 < 8; r8++) st[r8] = acc[r8] + bt[r8 + 8 * hi];
    float4* dst = (float4*)(y1 + (((size_t)b * IMGSZ + oy) * IMGSZ + ox) * CF + 8 * hi);
    dst[0] = make_float4(st[0], st[1], st[2], st[3]);
    dst[1] = make_float4(st[4], st[5], st[6], st[7]);
}

// ---------------------------------------------------------------------------
// BN stats over channel-last tensor; emits fused affine sa = g*rstd,
// sb = be - mean*g*rstd so convs apply x*sa + sb then leaky.
// ---------------------------------------------------------------------------
__global__ void vr_bnstats(const float* __restrict__ x, int nch,
                           const float* __restrict__ g, const float* __restrict__ be,
                           float* __restrict__ sa, float* __restrict__ sb)
{
    int ch = blockIdx.x;
    int tid = threadIdx.x;
    const int N = NB * IMGSZ * IMGSZ;
    float s = 0.f, ss = 0.f;
    for (int i = tid; i < N; i += blockDim.x) {
        float v = x[(size_t)i * nch + ch];
        s += v; ss += v * v;
    }
    __shared__ float sh1[256], sh2[256];
    sh1[tid] = s; sh2[tid] = ss;
    __syncthreads();
    for (int off = 128; off > 0; off >>= 1) {
        if (tid < off) { sh1[tid] += sh1[tid + off]; sh2[tid] += sh2[tid + off]; }
        __syncthreads();
    }
    if (tid == 0) {
        float m = sh1[0] / (float)N;
        float v = sh2[0] / (float)N - m * m;
        float rs = rsqrtf(v + 1e-5f) * g[ch];
        sa[ch] = rs;
        sb[ch] = be[ch] - m * rs;
    }
}

// ---------------------------------------------------------------------------
// Kernel 3: conv 16->8 (5x5 pad 2), fused BN1-affine + leaky on input loads.
// ---------------------------------------------------------------------------
__global__ void vr_conv1(const float* __restrict__ y1, const _Float16* __restrict__ aP1,
                         const float* __restrict__ b1, const float* __restrict__ sa1,
                         const float* __restrict__ sb1, float* __restrict__ y2)
{
    int wave = (blockIdx.x * blockDim.x + threadIdx.x) >> 5;
    int lane = threadIdx.x & 31;
    int b = wave >> 10;
    int rem = wave & 1023;
    int oy = rem >> 3, tx = rem & 7;
    int hi = lane >> 4, n = lane & 15;
    int ox = tx * 16 + n;

    float sa[16], sb[16];
    #pragma unroll
    for (int j = 0; j < 16; j++) { sa[j] = sa1[j]; sb[j] = sb1[j]; }

    const v16h* aFrag = (const v16h*)aP1 + lane;

    v8f acc = {0.f,0.f,0.f,0.f,0.f,0.f,0.f,0.f};
    #pragma unroll
    for (int c = 0; c < 13; c++) {
        v16h a = aFrag[c * 32];
        int t = 2 * c + hi;
        int tc_ = (t < 25) ? t : 24;
        int iy = oy + tc_ / 5 - 2, ix = ox + (tc_ % 5) - 2;
        float msk = ((t < 25) && (iy >= 0) && (iy < IMGSZ) && (ix >= 0) && (ix < IMGSZ)) ? 1.f : 0.f;
        int iyc = min(max(iy, 0), IMGSZ - 1);
        int ixc = min(max(ix, 0), IMGSZ - 1);
        const float4* p = (const float4*)(y1 + (((size_t)b * IMGSZ + iyc) * IMGSZ + ixc) * CF);
        float4 q0 = p[0], q1 = p[1], q2 = p[2], q3 = p[3];
        float tmp[16] = {q0.x,q0.y,q0.z,q0.w, q1.x,q1.y,q1.z,q1.w,
                         q2.x,q2.y,q2.z,q2.w, q3.x,q3.y,q3.z,q3.w};
        v16h bb;
        #pragma unroll
        for (int j = 0; j < 16; j++) {
            float v = tmp[j] * sa[j] + sb[j];
            v = (v >= 0.f) ? v : 0.01f * v;
            bb[j] = (_Float16)(v * msk);
        }
        acc = wmma16(a, bb, acc);
    }
    if (hi == 0) {
        float st[8];
        #pragma unroll
        for (int r8 = 0; r8 < 8; r8++) st[r8] = acc[r8] + b1[r8];
        float4* dst = (float4*)(y2 + (((size_t)b * IMGSZ + oy) * IMGSZ + ox) * 8);
        dst[0] = make_float4(st[0], st[1], st[2], st[3]);
        dst[1] = make_float4(st[4], st[5], st[6], st[7]);
    }
}

// ---------------------------------------------------------------------------
// Kernel 4: conv 8->3 (5x5 pad 2), fused BN2-affine + leaky, ReLU output.
// ---------------------------------------------------------------------------
__global__ void vr_conv2(const float* __restrict__ y2, const _Float16* __restrict__ aP2,
                         const float* __restrict__ b2, const float* __restrict__ sa2,
                         const float* __restrict__ sb2, float* __restrict__ out)
{
    int wave = (blockIdx.x * blockDim.x + threadIdx.x) >> 5;
    int lane = threadIdx.x & 31;
    int b = wave >> 10;
    int rem = wave & 1023;
    int oy = rem >> 3, tx = rem & 7;
    int hi = lane >> 4, n = lane & 15;
    int ox = tx * 16 + n;

    float sa[8], sb[8];
    #pragma unroll
    for (int j = 0; j < 8; j++) { sa[j] = sa2[j]; sb[j] = sb2[j]; }

    const v16h* aFrag = (const v16h*)aP2 + lane;

    v8f acc = {0.f,0.f,0.f,0.f,0.f,0.f,0.f,0.f};
    #pragma unroll
    for (int c = 0; c < 7; c++) {
        v16h a = aFrag[c * 32];
        v16h bb;
        #pragma unroll
        for (int jb = 0; jb < 2; jb++) {
            int t = 4 * c + jb + 2 * hi;
            int tc_ = (t < 25) ? t : 24;
            int iy = oy + tc_ / 5 - 2, ix = ox + (tc_ % 5) - 2;
            float msk = ((t < 25) && (iy >= 0) && (iy < IMGSZ) && (ix >= 0) && (ix < IMGSZ)) ? 1.f : 0.f;
            int iyc = min(max(iy, 0), IMGSZ - 1);
            int ixc = min(max(ix, 0), IMGSZ - 1);
            const float4* p = (const float4*)(y2 + (((size_t)b * IMGSZ + iyc) * IMGSZ + ixc) * 8);
            float4 q0 = p[0], q1 = p[1];
            float tmp[8] = {q0.x,q0.y,q0.z,q0.w, q1.x,q1.y,q1.z,q1.w};
            #pragma unroll
            for (int j = 0; j < 8; j++) {
                float v = tmp[j] * sa[j] + sb[j];
                v = (v >= 0.f) ? v : 0.01f * v;
                bb[jb * 8 + j] = (_Float16)(v * msk);
            }
        }
        acc = wmma16(a, bb, acc);
    }
    if (hi == 0) {
        #pragma unroll
        for (int r8 = 0; r8 < 3; r8++) {
            float v = acc[r8] + b2[r8];
            out[(((size_t)b * 3 + r8) * IMGSZ + oy) * IMGSZ + ox] = fmaxf(v, 0.f);
        }
    }
}

// ---------------------------------------------------------------------------
// Kernel 5: bilinear (half-pixel) 64->128 upsample of silhouette.
// ---------------------------------------------------------------------------
__global__ void vr_silup(const float* __restrict__ sil, float* __restrict__ out)
{
    int tid = blockIdx.x * blockDim.x + threadIdx.x;
    if (tid >= NB * IMGSZ * IMGSZ) return;
    int b = tid >> 14;
    int rem = tid & 16383;
    int oy = rem >> 7, ox = rem & 127;
    float sx = fminf(fmaxf((ox + 0.5f) * 0.5f - 0.5f, 0.f), 63.f);
    float sy = fminf(fmaxf((oy + 0.5f) * 0.5f - 0.5f, 0.f), 63.f);
    int x0 = (int)floorf(sx), y0 = (int)floorf(sy);
    int x1 = min(x0 + 1, 63), y1 = min(y0 + 1, 63);
    float fx = sx - (float)x0, fy = sy - (float)y0;
    const float* sb = sil + (size_t)b * HWH * HWH;
    float v00 = sb[y0 * HWH + x0], v01 = sb[y0 * HWH + x1];
    float v10 = sb[y1 * HWH + x0], v11 = sb[y1 * HWH + x1];
    out[tid] = (1.f - fy) * ((1.f - fx) * v00 + fx * v01)
             +        fy  * ((1.f - fx) * v10 + fx * v11);
}

extern "C" void kernel_launch(void* const* d_in, const int* in_sizes, int n_in,
                              void* d_out, int out_size, void* d_ws, size_t ws_size,
                              hipStream_t stream) {
    const float* R    = (const float*)d_in[0];
    const float* T    = (const float*)d_in[1];
    const float* Kc   = (const float*)d_in[2];
    const float* feat = (const float*)d_in[3];
    const float* dens = (const float*)d_in[4];
    const float* wt   = (const float*)d_in[5];
    const float* bt   = (const float*)d_in[6];
    const float* g1   = (const float*)d_in[7];
    const float* be1  = (const float*)d_in[8];
    const float* w1   = (const float*)d_in[9];
    const float* b1   = (const float*)d_in[10];
    const float* g2   = (const float*)d_in[11];
    const float* be2  = (const float*)d_in[12];
    const float* w2   = (const float*)d_in[13];
    const float* b2   = (const float*)d_in[14];

    float* ws  = (float*)d_ws;
    float* img = ws;                       // (8,64,64,16)  = 524288
    float* sil = img + 524288;             // (8,64,64)     = 32768
    float* y1  = sil + 32768;              // (8,128,128,16)= 2097152
    float* y2  = y1 + 2097152;             // (8,128,128,8) = 1048576
    float* sa1 = y2 + 1048576;             // 16
    float* sb1 = sa1 + 16;                 // 16
    float* sa2 = sb1 + 16;                 // 8
    float* sb2 = sa2 + 8;                  // 8
    _Float16* aT  = (_Float16*)(sb2 + 8);  // 4*5*32*16  = 10240 halfs (32B aligned)
    _Float16* aP1 = aT + 10240;            // 13*32*16   = 6656 halfs
    _Float16* aP2 = aP1 + 6656;            // 7*32*16    = 3584 halfs
    float* out = (float*)d_out;

    vr_pack_wt<<<3, 256, 0, stream>>>(wt, aT);
    vr_pack_w1<<<2, 256, 0, stream>>>(w1, aP1);
    vr_pack_w2<<<1, 256, 0, stream>>>(w2, aP2);
    vr_render <<<128, 256, 0, stream>>>(R, T, Kc, feat, dens, img, sil);
    vr_convt  <<<1024, 256, 0, stream>>>(img, aT, bt, y1);
    vr_bnstats<<<16, 256, 0, stream>>>(y1, 16, g1, be1, sa1, sb1);
    vr_conv1  <<<1024, 256, 0, stream>>>(y1, aP1, b1, sa1, sb1, y2);
    vr_bnstats<<<8, 256, 0, stream>>>(y2, 8, g2, be2, sa2, sb2);
    vr_conv2  <<<1024, 256, 0, stream>>>(y2, aP2, b2, sa2, sb2, out);
    vr_silup  <<<512, 256, 0, stream>>>(sil, out + (size_t)NB * 3 * IMGSZ * IMGSZ);
}
